// EdgeNetVAE_7456063226141
// MI455X (gfx1250) — matrix operand order repaired and checked by reading, not compiled, not run on hardware
//
#include <hip/hip_runtime.h>
#include <hip/hip_bf16.h>

// ---------------------------------------------------------------------------
// EdgeConv VAE on gfx1250 (MI455X).  Per-edge MLPs run on v_wmma_f32_16x16x32_f16
// (wave32, M=16 edges, N=16 out-channels, K=32 features, f32 accum).
// Layer-to-layer re-layout uses a channel-pair-permuted LDS tile so stores are
// ds_store_b32 and A-fragment reads are ds_load_b128.
// ---------------------------------------------------------------------------

typedef __attribute__((ext_vector_type(16))) _Float16 v16h;
typedef __attribute__((ext_vector_type(8)))  _Float16 v8h;
typedef __attribute__((ext_vector_type(2)))  _Float16 v2h;
typedef __attribute__((ext_vector_type(8)))  float    v8f;

union AF { v16h v; v8h g[2]; _Float16 h[16]; };
union CF { v8f  v; float    f[8];  };

#define WAVES 8          // 256-thread block = 8 wave32
#define TPB   256
#define LROW  40         // LDS row stride in halves: 80B, 16B-aligned, bank-spread

// ISA 7.12.2: 16-bit A 16x32.  lane<16: v0..3 hold K0..7, v4..7 hold K16..23.
// lane>=16: +8 on the low group (K8..15), +8 on the high group (K24..31).
__device__ __forceinline__ int a_kidx(int lane, int i) {
    return ((i & 8) ? 16 : 0) + ((lane & 16) ? 8 : 0) + (i & 7);
}
// Contiguous k order (unused here): lanes 0-15 hold K=0..15, lanes 16-31 hold K=16..31; col = lane&15.
__device__ __forceinline__ int b_kidx(int lane, int i) {
    return ((lane & 16) ? 16 : 0) + i;
}
// Virtual (packed-pair) channel v -> original channel: even v -> v/2 (lo frag),
// odd v -> 16 + v/2 (hi frag).  Inverse of the packed ds_store_b32 layout.
__device__ __forceinline__ int origk(int v) {
    return (v & 1) ? 16 + (v >> 1) : (v >> 1);
}

__device__ __forceinline__ v8f wmma16(v16h a, v16h b, v8f c) {
    return __builtin_amdgcn_wmma_f32_16x16x32_f16(
        /*neg_a=*/false, a, /*neg_b=*/false, b,
        /*c_mod=*/(short)0, c, /*reuse_a=*/false, /*reuse_b=*/false);
}

// ---------------------------------------------------------------------------
// BatchNorm statistics: sum / sumsq per feature (D=4), wave-reduced + atomics.
// ---------------------------------------------------------------------------
__global__ void __launch_bounds__(TPB)
bn_reduce_kernel(const float* __restrict__ x, float* __restrict__ stat, int N) {
    const float4* __restrict__ x4 = (const float4*)x;
    float s[4] = {0.f, 0.f, 0.f, 0.f};
    float q[4] = {0.f, 0.f, 0.f, 0.f};
    for (int n = blockIdx.x * blockDim.x + threadIdx.x; n < N;
         n += gridDim.x * blockDim.x) {
        float4 v = x4[n];
        s[0] += v.x; q[0] += v.x * v.x;
        s[1] += v.y; q[1] += v.y * v.y;
        s[2] += v.z; q[2] += v.z * v.z;
        s[3] += v.w; q[3] += v.w * v.w;
    }
    #pragma unroll
    for (int off = 16; off > 0; off >>= 1) {
        #pragma unroll
        for (int k = 0; k < 4; ++k) {
            s[k] += __shfl_down(s[k], off, 32);
            q[k] += __shfl_down(q[k], off, 32);
        }
    }
    if ((threadIdx.x & 31) == 0) {
        #pragma unroll
        for (int k = 0; k < 4; ++k) {
            atomicAdd(&stat[k],     s[k]);
            atomicAdd(&stat[4 + k], q[k]);
        }
    }
}

__global__ void bn_finalize_kernel(const float* __restrict__ stat,
                                   const float* __restrict__ gamma,
                                   const float* __restrict__ beta,
                                   float* __restrict__ bnsc, int N) {
    int k = threadIdx.x;
    if (k >= 4) return;
    float invN  = 1.0f / (float)N;
    float mean  = stat[k] * invN;
    float var   = stat[4 + k] * invN - mean * mean;
    float scale = gamma[k] * rsqrtf(var + 1e-5f);
    bnsc[k]     = scale;
    bnsc[4 + k] = beta[k] - mean * scale;
}

// ---------------------------------------------------------------------------
// Node in-degree (float, shared by both EdgeConvs; mean = sum / max(cnt,1)).
// ---------------------------------------------------------------------------
__global__ void __launch_bounds__(TPB)
count_kernel(const long long* __restrict__ dst, float* __restrict__ cnt,
             long long E) {
    for (long long e = blockIdx.x * (long long)blockDim.x + threadIdx.x; e < E;
         e += (long long)gridDim.x * blockDim.x) {
        atomicAdd(&cnt[dst[e]], 1.0f);
    }
}

// ---------------------------------------------------------------------------
// Encoder EdgeConv: e = [x_i, x_j - x_i] (8) -> 32 (relu) -> 32 (relu),
// scatter-add into hsum[dst][32].  BN affine fused into the gather.
// ---------------------------------------------------------------------------
__global__ void __launch_bounds__(TPB)
enc_edge_kernel(const float* __restrict__ x,
                const long long* __restrict__ src,
                const long long* __restrict__ dst,
                const float* __restrict__ bnsc,
                const float* __restrict__ w1, const float* __restrict__ b1,
                const float* __restrict__ w2, const float* __restrict__ b2,
                float* __restrict__ hsum, long long E) {
    __shared__ __align__(16) _Float16 sH[WAVES][16][LROW];

    const float4* __restrict__ x4 = (const float4*)x;

    const int lane = threadIdx.x & 31;
    const int wave = threadIdx.x >> 5;
    const int col  = lane & 15;
    const int hi   = (lane & 16) ? 1 : 0;

    float sc[4], sh[4];
    #pragma unroll
    for (int k = 0; k < 4; ++k) { sc[k] = bnsc[k]; sh[k] = bnsc[4 + k]; }

    // Constant weight B-fragments, resident in VGPRs across the edge loop.
    // Layer 1: natural K (raw edge features).  Layer 2: K is the packed-pair
    // virtual channel order produced by the ds_store_b32 staging.
    AF b1lo, b1hi, b2lo, b2hi;
    #pragma unroll
    for (int i = 0; i < 16; ++i) {
        int K  = a_kidx(lane, i);
        int ok = origk(K);
        b1lo.h[i] = (K < 8) ? (_Float16)w1[col * 8 + K]        : (_Float16)0.f;
        b1hi.h[i] = (K < 8) ? (_Float16)w1[(16 + col) * 8 + K] : (_Float16)0.f;
        b2lo.h[i] = (_Float16)w2[col * 32 + ok];
        b2hi.h[i] = (_Float16)w2[(16 + col) * 32 + ok];
    }
    const float bias1lo = b1[col], bias1hi = b1[16 + col];
    const float bias2lo = b2[col], bias2hi = b2[16 + col];

    const long long nTiles  = (E + 31) / 32;
    const long long tStride = (long long)gridDim.x * WAVES;
    const int base0 = hi ? 8 : 0;       // A-fragment b128 chunk offsets
    const int base1 = 16 + base0;

    for (long long tile = (long long)blockIdx.x * WAVES + wave; tile < nTiles;
         tile += tStride) {
        long long e = tile * 32 + lane;
        // prefetch next tile's edge indices into L2 (global_prefetch_b8)
        __builtin_prefetch(&src[e + tStride * 32], 0, 0);
        __builtin_prefetch(&dst[e + tStride * 32], 0, 0);

        float feat[8];
        #pragma unroll
        for (int k = 0; k < 8; ++k) feat[k] = 0.f;
        int dIdx = -1;
        if (e < E) {
            long long si = src[e], di = dst[e];
            dIdx = (int)di;
            float4 xiv = x4[di];                        // global_load_b128
            float4 xjv = x4[si];
            float xi[4] = {xiv.x, xiv.y, xiv.z, xiv.w};
            float xj[4] = {xjv.x, xjv.y, xjv.z, xjv.w};
            #pragma unroll
            for (int k = 0; k < 4; ++k) {
                float a = xi[k] * sc[k] + sh[k];
                float b = xj[k] * sc[k] + sh[k];
                feat[k]     = a;
                feat[4 + k] = b - a;
            }
        }

        #pragma unroll
        for (int t = 0; t < 2; ++t) {           // two 16-edge sub-tiles
            // A fragment: rows live in lanes 0-15; K>=8 is zero padding.
            AF a;
            #pragma unroll
            for (int i = 0; i < 16; ++i) a.h[i] = (_Float16)0.f;
            #pragma unroll
            for (int i = 0; i < 8; ++i) {
                float f = t ? __shfl(feat[i], col + 16, 32) : feat[i];
                if (!hi) a.h[i] = (_Float16)f;
            }
            // Layer 1: 8->32 (two WMMAs for the 32 output channels)
            v8f z = {};
            CF c0, c1;
            c0.v = wmma16(a.v, b1lo.v, z);
            c1.v = wmma16(a.v, b1hi.v, z);
            // bias + relu; packed channel-pair store (ds_store_b32)
            #pragma unroll
            for (int r = 0; r < 8; ++r) {
                int m = r + 8 * hi;
                v2h p = {(_Float16)fmaxf(c0.f[r] + bias1lo, 0.f),
                         (_Float16)fmaxf(c1.f[r] + bias1hi, 0.f)};
                *(v2h*)&sH[wave][m][2 * col] = p;
            }
            // A2 fragment: two ds_load_b128 per lane
            AF a2;
            a2.g[0] = *(const v8h*)&sH[wave][col][base0];
            a2.g[1] = *(const v8h*)&sH[wave][col][base1];
            // Layer 2: 32->32 (K=32 exact fit, permuted K order in B)
            CF d0, d1;
            d0.v = wmma16(a2.v, b2lo.v, z);
            d1.v = wmma16(a2.v, b2hi.v, z);
            // bias + relu + scatter-add (mean numerator)
            #pragma unroll
            for (int r = 0; r < 8; ++r) {
                int dn = __shfl(dIdx, 16 * t + r + 8 * hi, 32);
                if (dn >= 0) {
                    atomicAdd(&hsum[(long long)dn * 32 + col],
                              fmaxf(d0.f[r] + bias2lo, 0.f));
                    atomicAdd(&hsum[(long long)dn * 32 + 16 + col],
                              fmaxf(d1.f[r] + bias2hi, 0.f));
                }
            }
        }
    }
}

// ---------------------------------------------------------------------------
// Per-node: h = hsum/cnt, mu/logvar GEMV (32->2), reparameterize z.
// ---------------------------------------------------------------------------
__global__ void __launch_bounds__(TPB)
node_mid_kernel(const float* __restrict__ hsum, const float* __restrict__ cnt,
                const float* __restrict__ mu_w, const float* __restrict__ mu_b,
                const float* __restrict__ var_w, const float* __restrict__ var_b,
                const float* __restrict__ eps,
                float* __restrict__ out_mu, float* __restrict__ out_lv,
                float* __restrict__ zbuf, int N) {
    int n = blockIdx.x * blockDim.x + threadIdx.x;
    if (n >= N) return;
    float inv = 1.0f / fmaxf(cnt[n], 1.0f);
    const float4* __restrict__ h4 = (const float4*)&hsum[(long long)n * 32];
    float h[32];
    #pragma unroll
    for (int q = 0; q < 8; ++q) {
        float4 v = h4[q];
        h[q * 4 + 0] = v.x * inv; h[q * 4 + 1] = v.y * inv;
        h[q * 4 + 2] = v.z * inv; h[q * 4 + 3] = v.w * inv;
    }
    #pragma unroll
    for (int j = 0; j < 2; ++j) {
        float mu = mu_b[j], lv = var_b[j];
        #pragma unroll
        for (int k = 0; k < 32; ++k) {
            mu = fmaf(h[k], mu_w[j * 32 + k], mu);
            lv = fmaf(h[k], var_w[j * 32 + k], lv);
        }
        float zz = mu + eps[(long long)n * 2 + j] * expf(0.5f * lv);
        out_mu[(long long)n * 2 + j] = mu;
        out_lv[(long long)n * 2 + j] = lv;
        zbuf[(long long)n * 2 + j]   = zz;
    }
}

// ---------------------------------------------------------------------------
// Decoder EdgeConv: e = [z_i, z_j - z_i] (4) -> 32 (relu) -> 32 (relu) -> 4,
// scatter-add into osum[dst][4].
// ---------------------------------------------------------------------------
__global__ void __launch_bounds__(TPB)
dec_edge_kernel(const float* __restrict__ zin,
                const long long* __restrict__ src,
                const long long* __restrict__ dst,
                const float* __restrict__ w1, const float* __restrict__ b1,
                const float* __restrict__ w2, const float* __restrict__ b2,
                const float* __restrict__ w3, const float* __restrict__ b3,
                float* __restrict__ osum, long long E) {
    __shared__ __align__(16) _Float16 sH[WAVES][16][LROW];

    const float2* __restrict__ z2 = (const float2*)zin;

    const int lane = threadIdx.x & 31;
    const int wave = threadIdx.x >> 5;
    const int col  = lane & 15;
    const int hi   = (lane & 16) ? 1 : 0;

    AF b1lo, b1hi, b2lo, b2hi, b3f;
    #pragma unroll
    for (int i = 0; i < 16; ++i) {
        int K  = a_kidx(lane, i);
        int ok = origk(K);
        b1lo.h[i] = (K < 4) ? (_Float16)w1[col * 4 + K]        : (_Float16)0.f;
        b1hi.h[i] = (K < 4) ? (_Float16)w1[(16 + col) * 4 + K] : (_Float16)0.f;
        b2lo.h[i] = (_Float16)w2[col * 32 + ok];
        b2hi.h[i] = (_Float16)w2[(16 + col) * 32 + ok];
        b3f.h[i]  = (col < 4) ? (_Float16)w3[col * 32 + ok]    : (_Float16)0.f;
    }
    const float bias1lo = b1[col], bias1hi = b1[16 + col];
    const float bias2lo = b2[col], bias2hi = b2[16 + col];
    const float bias3   = (col < 4) ? b3[col] : 0.f;

    const long long nTiles  = (E + 31) / 32;
    const long long tStride = (long long)gridDim.x * WAVES;
    const int base0 = hi ? 8 : 0;
    const int base1 = 16 + base0;

    for (long long tile = (long long)blockIdx.x * WAVES + wave; tile < nTiles;
         tile += tStride) {
        long long e = tile * 32 + lane;
        __builtin_prefetch(&src[e + tStride * 32], 0, 0);
        __builtin_prefetch(&dst[e + tStride * 32], 0, 0);

        float feat[8];
        #pragma unroll
        for (int k = 0; k < 8; ++k) feat[k] = 0.f;
        int dIdx = -1;
        if (e < E) {
            long long si = src[e], di = dst[e];
            dIdx = (int)di;
            float2 zi = z2[di];                         // global_load_b64
            float2 zj = z2[si];
            feat[0] = zi.x;        feat[1] = zi.y;
            feat[2] = zj.x - zi.x; feat[3] = zj.y - zi.y;
        }

        #pragma unroll
        for (int t = 0; t < 2; ++t) {
            AF a;
            #pragma unroll
            for (int i = 0; i < 16; ++i) a.h[i] = (_Float16)0.f;
            #pragma unroll
            for (int i = 0; i < 4; ++i) {
                float f = t ? __shfl(feat[i], col + 16, 32) : feat[i];
                if (!hi) a.h[i] = (_Float16)f;
            }
            v8f z = {};
            // Layer 1: 4->32
            CF c0, c1;
            c0.v = wmma16(a.v, b1lo.v, z);
            c1.v = wmma16(a.v, b1hi.v, z);
            #pragma unroll
            for (int r = 0; r < 8; ++r) {
                int m = r + 8 * hi;
                v2h p = {(_Float16)fmaxf(c0.f[r] + bias1lo, 0.f),
                         (_Float16)fmaxf(c1.f[r] + bias1hi, 0.f)};
                *(v2h*)&sH[wave][m][2 * col] = p;
            }
            AF a2;
            a2.g[0] = *(const v8h*)&sH[wave][col][base0];
            a2.g[1] = *(const v8h*)&sH[wave][col][base1];
            // Layer 2: 32->32
            CF d0, d1;
            d0.v = wmma16(a2.v, b2lo.v, z);
            d1.v = wmma16(a2.v, b2hi.v, z);
            #pragma unroll
            for (int r = 0; r < 8; ++r) {
                int m = r + 8 * hi;
                v2h p = {(_Float16)fmaxf(d0.f[r] + bias2lo, 0.f),
                         (_Float16)fmaxf(d1.f[r] + bias2hi, 0.f)};
                *(v2h*)&sH[wave][m][2 * col] = p;
            }
            AF a3;
            a3.g[0] = *(const v8h*)&sH[wave][col][base0];
            a3.g[1] = *(const v8h*)&sH[wave][col][base1];
            // Layer 3: 32->4 (single WMMA, cols 0..3 valid, no relu)
            CF d3;
            d3.v = wmma16(a3.v, b3f.v, z);
            #pragma unroll
            for (int r = 0; r < 8; ++r) {
                int dn = __shfl(dIdx, 16 * t + r + 8 * hi, 32);
                if (dn >= 0 && col < 4)
                    atomicAdd(&osum[(long long)dn * 4 + col], d3.f[r] + bias3);
            }
        }
    }
}

// ---------------------------------------------------------------------------
// out = osum / max(cnt, 1)
// ---------------------------------------------------------------------------
__global__ void __launch_bounds__(TPB)
out_final_kernel(const float* __restrict__ osum, const float* __restrict__ cnt,
                 float* __restrict__ out, int N) {
    int n = blockIdx.x * blockDim.x + threadIdx.x;
    if (n >= N) return;
    float inv = 1.0f / fmaxf(cnt[n], 1.0f);
    float4 v = ((const float4*)osum)[n];
    float4 o = {v.x * inv, v.y * inv, v.z * inv, v.w * inv};
    ((float4*)out)[n] = o;
}

// ---------------------------------------------------------------------------
extern "C" void kernel_launch(void* const* d_in, const int* in_sizes, int n_in,
                              void* d_out, int out_size, void* d_ws,
                              size_t ws_size, hipStream_t stream) {
    const float*     x     = (const float*)d_in[0];
    const float*     eps   = (const float*)d_in[1];
    const float*     gam   = (const float*)d_in[2];
    const float*     bet   = (const float*)d_in[3];
    const float*     ew1   = (const float*)d_in[4];
    const float*     eb1   = (const float*)d_in[5];
    const float*     ew2   = (const float*)d_in[6];
    const float*     eb2   = (const float*)d_in[7];
    const float*     mu_w  = (const float*)d_in[8];
    const float*     mu_b  = (const float*)d_in[9];
    const float*     va_w  = (const float*)d_in[10];
    const float*     va_b  = (const float*)d_in[11];
    const float*     dw1   = (const float*)d_in[12];
    const float*     db1   = (const float*)d_in[13];
    const float*     dw2   = (const float*)d_in[14];
    const float*     db2   = (const float*)d_in[15];
    const float*     dw3   = (const float*)d_in[16];
    const float*     db3   = (const float*)d_in[17];
    const long long* eidx  = (const long long*)d_in[18];

    const int       N   = in_sizes[0] / 4;
    const long long E   = (long long)in_sizes[18] / 2;
    const long long* src = eidx;
    const long long* dst = eidx + E;

    // workspace layout (floats), every region 16-float (64B) aligned
    auto rnd = [](long long v) { return (v + 15) & ~15LL; };
    float* wsf    = (float*)d_ws;
    float* bnstat = wsf;                               // 8  (+8 bnsc)
    float* bnsc   = wsf + 8;
    float* cnt    = wsf + 16;                          // N
    float* hsum   = cnt  + rnd(N);                     // 32N
    float* zbuf   = hsum + rnd(32LL * N);              // 2N
    float* osum   = zbuf + rnd(2LL * N);               // 4N

    float* out    = (float*)d_out;                     // [N*4]
    float* out_mu = out + 4LL * N;                     // [N*2]
    float* out_lv = out + 6LL * N;                     // [N*2]

    // zero accumulators (bnstat; cnt..hsum contiguous; osum)
    hipMemsetAsync(bnstat, 0, 16 * sizeof(float), stream);
    hipMemsetAsync(cnt, 0, (size_t)(rnd(N) + rnd(32LL * N)) * sizeof(float),
                   stream);
    hipMemsetAsync(osum, 0, (size_t)(4LL * N) * sizeof(float), stream);

    // BatchNorm statistics + affine
    bn_reduce_kernel<<<512, TPB, 0, stream>>>(x, bnstat, N);
    bn_finalize_kernel<<<1, 4, 0, stream>>>(bnstat, gam, bet, bnsc, N);

    // in-degree counts
    count_kernel<<<4096, TPB, 0, stream>>>(dst, cnt, E);

    // encoder EdgeConv (WMMA)
    enc_edge_kernel<<<1024, TPB, 0, stream>>>(x, src, dst, bnsc,
                                              ew1, eb1, ew2, eb2, hsum, E);

    // node head: mean-aggregate, mu/logvar, reparameterize
    node_mid_kernel<<<(N + TPB - 1) / TPB, TPB, 0, stream>>>(
        hsum, cnt, mu_w, mu_b, va_w, va_b, eps, out_mu, out_lv, zbuf, N);

    // decoder EdgeConv (WMMA)
    dec_edge_kernel<<<1024, TPB, 0, stream>>>(zbuf, src, dst,
                                              dw1, db1, dw2, db2, dw3, db3,
                                              osum, E);

    // mean-divide into output
    out_final_kernel<<<(N + TPB - 1) / TPB, TPB, 0, stream>>>(osum, cnt,
                                                              out, N);
}
